// DCAM_ADB_TRI_44933947851413
// MI455X (gfx1250) — matrix operand order repaired and checked by reading, compiled, not run on hardware
//
#include <hip/hip_runtime.h>
#include <hip/hip_bf16.h>

typedef __attribute__((ext_vector_type(16))) _Float16 v16h;
typedef __attribute__((ext_vector_type(8)))  _Float16 v8h;
typedef __attribute__((ext_vector_type(8)))  float    v8f;

#define WS   7
#define NTOK 49
#define DIM  192
#define NH   6
#define HD   32
#define BW   2048
#define NPAD 64

// ---------------------------------------------------------------------------
// Fragment loaders: contiguous k order (k = 16*(lane/16) + s), same in A and B
// ---------------------------------------------------------------------------

// A-matrix 16x32 f16: lanes 0-15 hold K=[0..15]; lanes 16-31 hold
// K=[16..31]; lane%16 = M row. Source: row-major halfs, stride in halfs.
__device__ __forceinline__ v16h load_a_frag(const _Float16* base, int stride,
                                            int row_base, int k_base, int lid) {
    int r = lid & 15;
    int hb = (lid >> 4) * 16;
    const _Float16* p = base + (row_base + r) * stride + k_base;
    v8h lo = *(const v8h*)(p + hb);          // K = k_base + hb .. +8
    v8h hi = *(const v8h*)(p + hb + 8);      // K = k_base + hb + 8 .. +8
    v16h a;
#pragma unroll
    for (int i = 0; i < 8; i++) { a[i] = lo[i]; a[8 + i] = hi[i]; }
    return a;
}

// B-matrix 32x16 f16: lane%16 = N column; lanes 0-15 hold K=0..15, lanes 16-31
// hold K=16..31 (16 consecutive K per lane). Source W is "N-major" row-major
// with K contiguous: B[K][N] = W[N][K]  -> one 32B contiguous load per lane.
__device__ __forceinline__ v16h load_b_frag(const _Float16* base, int stride,
                                            int n_base, int k_base, int lid) {
    int c = lid & 15;
    int kh = (lid >> 4) * 16;
    return *(const v16h*)(base + (n_base + c) * stride + k_base + kh);
}

__device__ __forceinline__ v8f wmma16(v16h a, v16h b, v8f c) {
    return __builtin_amdgcn_wmma_f32_16x16x32_f16(false, a, false, b,
                                                  (short)0, c, false, false);
}

// ---------------------------------------------------------------------------
// Pre-kernel: f32->f16 weight conversion + padded bias gather [NH][64][64]
// ---------------------------------------------------------------------------
__global__ void winattn_prep(const float* __restrict__ qkv_w,
                             const float* __restrict__ proj_w,
                             const float* __restrict__ bias_table,
                             const int*   __restrict__ rel_index,
                             _Float16* __restrict__ qkvw_h,
                             _Float16* __restrict__ projw_h,
                             float*    __restrict__ bias_full) {
    int i = blockIdx.x * blockDim.x + threadIdx.x;
    if (i < 3 * DIM * DIM) qkvw_h[i] = (_Float16)qkv_w[i];
    if (i < DIM * DIM)     projw_h[i] = (_Float16)proj_w[i];
    if (i < NH * NPAD * NPAD) {
        int h = i / (NPAD * NPAD);
        int rem = i % (NPAD * NPAD);
        int n = rem / NPAD, m = rem % NPAD;
        float v = 0.f;
        if (n < NTOK && m < NTOK)
            v = bias_table[rel_index[n * NTOK + m] * NH + h];
        bias_full[i] = v;
    }
}

// ---------------------------------------------------------------------------
// Phase-1 segment: one of q/k/v (WHICH = 0/1/2), compile-time store routing.
// 48 tiles (4 row-tiles x 12 col-tiles) split across 8 waves -> branch-free,
// EXEC stays all-ones around every WMMA.
// ---------------------------------------------------------------------------
template <int WHICH>
__device__ __forceinline__ void qkv_segment(const _Float16* __restrict__ xp_lds,
                                            const _Float16* __restrict__ qkvw,
                                            const float* __restrict__ qkv_b,
                                            _Float16* __restrict__ q_lds,
                                            _Float16* __restrict__ k_lds,
                                            _Float16* __restrict__ vT_lds,
                                            int wave, int lid, int lcol, int lhalf) {
    for (int t = wave; t < 48; t += 8) {
        int mt = t / 12, nt = t % 12;
        int oc = nt * 16 + lcol;                     // channel within segment
        const _Float16* wb = qkvw + (size_t)(WHICH * DIM + nt * 16) * DIM;
        v8f acc = {};
#pragma unroll
        for (int kt = 0; kt < 6; kt++) {
            if (kt < 5)                               // pull next B frag into WGP$
                __builtin_prefetch(wb + (size_t)lcol * DIM + (kt + 1) * 32, 0, 0);
            v16h a  = load_a_frag(xp_lds, DIM, mt * 16, kt * 32, lid);
            v16h bf = load_b_frag(wb,     DIM, 0,       kt * 32, lid);
            acc = wmma16(a, bf, acc);
        }
        float bia = qkv_b[WHICH * DIM + oc];
        int h = oc / HD, d = oc % HD;
        if (WHICH == 2) {
            // v^T[h][d][tok]: rows are consecutive per VGPR -> one b128 store
            v8h vv;
#pragma unroll
            for (int r = 0; r < 8; r++) vv[r] = (_Float16)(acc[r] + bia);
            *(v8h*)(&vT_lds[(h * HD + d) * NPAD + mt * 16 + lhalf * 8]) = vv;
        } else {
#pragma unroll
            for (int r = 0; r < 8; r++) {
                int row = mt * 16 + r + lhalf * 8;   // token (padded)
                float v = acc[r] + bia;
                if (WHICH == 0)
                    q_lds[(h * NPAD + row) * HD + d] =
                        (_Float16)(v * 0.17677669529663687f);  // 1/sqrt(32)
                else
                    k_lds[(h * NPAD + row) * HD + d] = (_Float16)v;
            }
        }
    }
}

// ---------------------------------------------------------------------------
// Main kernel: one workgroup (8 wave32) per window
// ---------------------------------------------------------------------------
__global__ __launch_bounds__(256)
void winattn_main(const float* __restrict__ x,
                  const float* __restrict__ qkv_b,
                  const float* __restrict__ proj_b,
                  const _Float16* __restrict__ qkvw,
                  const _Float16* __restrict__ projw,
                  const float* __restrict__ bias_full,
                  float* __restrict__ out) {
    // LDS: 49152 + 3*24576 + 24576 = 147456 B (<320KB WGP limit)
    __shared__ __align__(32) _Float16 xp_lds[NH * NPAD * NPAD]; // x (ph1) / P (ph2)
    __shared__ __align__(32) _Float16 q_lds [NH * NPAD * HD];   // q [h][tok][d]
    __shared__ __align__(32) _Float16 k_lds [NH * NPAD * HD];   // k [h][tok][d]
    __shared__ __align__(32) _Float16 vT_lds[NH * HD * NPAD];   // v^T [h][d][tok]
    __shared__ __align__(32) _Float16 ao_lds[NPAD * DIM];       // attn output rows

    const int b    = blockIdx.x;
    const int tid  = threadIdx.x;
    const int wave = tid >> 5;
    const int lid  = tid & 31;
    const int lcol = lid & 15;       // C/D: N column
    const int lhalf = lid >> 4;      // C/D: row = r + lhalf*8

    // ---- stage x window into LDS as f16, zero-pad rows 49..63 ----
    const float* xw = x + (size_t)b * NTOK * DIM;
    for (int idx = tid; idx < NPAD * DIM; idx += 256) {
        int row = idx / DIM, c = idx % DIM;
        float v = (row < NTOK) ? xw[row * DIM + c] : 0.f;
        xp_lds[idx] = (_Float16)v;
    }
    __syncthreads();

    // ---- Phase 1: QKV = x @ qkv_w^T + qkv_b (branch-free segments) ----
    qkv_segment<0>(xp_lds, qkvw, qkv_b, q_lds, k_lds, vT_lds, wave, lid, lcol, lhalf);
    qkv_segment<1>(xp_lds, qkvw, qkv_b, q_lds, k_lds, vT_lds, wave, lid, lcol, lhalf);
    qkv_segment<2>(xp_lds, qkvw, qkv_b, q_lds, k_lds, vT_lds, wave, lid, lcol, lhalf);
    __syncthreads();

    // ---- Phase 2: per (head, query-row-tile) attention ----
    _Float16* p_lds = xp_lds;            // reuse: [h][64][64]
    for (int u = wave; u < NH * 4; u += 8) {
        int h = u / 4, it = u % 4;
        const _Float16* qh = q_lds + h * NPAD * HD;
        const _Float16* kh = k_lds + h * NPAD * HD;
        const _Float16* vh = vT_lds + h * HD * NPAD;
        _Float16* ph = p_lds + h * NPAD * NPAD;

        // S tile row-block: 16 x 64, K = HD = 32 -> one WMMA per j-tile
        v16h aq = load_a_frag(qh, HD, it * 16, 0, lid);
        v8f s[4];
#pragma unroll
        for (int j = 0; j < 4; j++) {
            v8f z = {};
            v16h bk = load_b_frag(kh, HD, j * 16, 0, lid);
            s[j] = wmma16(aq, bk, z);
        }

        // bias + mask + row softmax (row lives in one 16-lane half)
        const float* bh = bias_full + h * NPAD * NPAD;
#pragma unroll
        for (int r = 0; r < 8; r++) {
            int row = it * 16 + r + lhalf * 8;
            float vals[4];
            float mx = -3.0e38f;
#pragma unroll
            for (int j = 0; j < 4; j++) {
                int col = j * 16 + lcol;
                float v = s[j][r] + bh[row * NPAD + col];
                if (col >= NTOK) v = -3.0e38f;
                vals[j] = v;
                mx = fmaxf(mx, v);
            }
            mx = fmaxf(mx, __shfl_xor(mx, 1, 32));
            mx = fmaxf(mx, __shfl_xor(mx, 2, 32));
            mx = fmaxf(mx, __shfl_xor(mx, 4, 32));
            mx = fmaxf(mx, __shfl_xor(mx, 8, 32));
            float sum = 0.f;
#pragma unroll
            for (int j = 0; j < 4; j++) {
                float e = __expf(vals[j] - mx);
                vals[j] = e;
                sum += e;
            }
            sum += __shfl_xor(sum, 1, 32);
            sum += __shfl_xor(sum, 2, 32);
            sum += __shfl_xor(sum, 4, 32);
            sum += __shfl_xor(sum, 8, 32);
            float inv = 1.f / sum;
#pragma unroll
            for (int j = 0; j < 4; j++)
                ph[row * NPAD + j * 16 + lcol] = (_Float16)(vals[j] * inv);
        }

        // out_i = P_i (16x64) @ v (64x32): 2 n-tiles x 2 k-steps
#pragma unroll
        for (int n2 = 0; n2 < 2; n2++) {
            v8f acc = {};
#pragma unroll
            for (int k2 = 0; k2 < 2; k2++) {
                v16h ap = load_a_frag(ph, NPAD, it * 16, k2 * 32, lid);
                v16h bv = load_b_frag(vh, NPAD, n2 * 16, k2 * 32, lid);
                acc = wmma16(ap, bv, acc);
            }
#pragma unroll
            for (int r = 0; r < 8; r++) {
                int row = it * 16 + r + lhalf * 8;
                int d = n2 * 16 + lcol;
                ao_lds[row * DIM + h * HD + d] = (_Float16)acc[r];
            }
        }
    }
    __syncthreads();

    // ---- Phase 3: projection ----
    float* ow = out + (size_t)b * NTOK * DIM;
    for (int t = wave; t < 4 * 12; t += 8) {
        int mt = t / 12, nt = t % 12;
        const _Float16* wb = projw + (size_t)(nt * 16) * DIM;
        v8f acc = {};
#pragma unroll
        for (int kt = 0; kt < 6; kt++) {
            if (kt < 5)
                __builtin_prefetch(wb + (size_t)lcol * DIM + (kt + 1) * 32, 0, 0);
            v16h a  = load_a_frag(ao_lds, DIM, mt * 16, kt * 32, lid);
            v16h bw = load_b_frag(wb,     DIM, 0,       kt * 32, lid);
            acc = wmma16(a, bw, acc);
        }
        int o = nt * 16 + lcol;
        float pb = proj_b[o];
#pragma unroll
        for (int r = 0; r < 8; r++) {
            int row = mt * 16 + r + lhalf * 8;
            if (row < NTOK) ow[row * DIM + o] = acc[r] + pb;
        }
    }
}

// ---------------------------------------------------------------------------
extern "C" void kernel_launch(void* const* d_in, const int* in_sizes, int n_in,
                              void* d_out, int out_size, void* d_ws, size_t ws_size,
                              hipStream_t stream) {
    const float* x          = (const float*)d_in[0];
    const float* qkv_w      = (const float*)d_in[1];
    const float* qkv_b      = (const float*)d_in[2];
    const float* proj_w     = (const float*)d_in[3];
    const float* proj_b     = (const float*)d_in[4];
    const float* bias_table = (const float*)d_in[5];
    const int*   rel_index  = (const int*)d_in[6];
    float* out = (float*)d_out;

    char* ws = (char*)d_ws;
    _Float16* qkvw_h = (_Float16*)ws;                               // 221184 B
    _Float16* projw_h = (_Float16*)(ws + 3 * DIM * DIM * 2);        //  73728 B
    float* bias_full = (float*)(ws + 3 * DIM * DIM * 2 + DIM * DIM * 2); // 98304 B

    int prep_elems = 3 * DIM * DIM;              // dominant range = 110592
    winattn_prep<<<(prep_elems + 255) / 256, 256, 0, stream>>>(
        qkv_w, proj_w, bias_table, rel_index, qkvw_h, projw_h, bias_full);

    winattn_main<<<BW, 256, 0, stream>>>(
        x, qkv_b, proj_b, qkvw_h, projw_h, bias_full, out);
}
